// UVMB_43224550866954
// MI455X (gfx1250) — matrix operand order repaired and checked by reading, not compiled, not run on hardware
//
#include <hip/hip_runtime.h>
#include <hip/hip_bf16.h>

// ---------- types ----------
typedef _Float16 v16h __attribute__((ext_vector_type(16)));
typedef _Float16 v8h  __attribute__((ext_vector_type(8)));
typedef float    v8f  __attribute__((ext_vector_type(8)));

#define B4   4
#define CH   64
#define HH   128
#define WW   128
#define LTOK 4096
#define BLT  (B4 * LTOK)     // 16384 tokens

// load a per-lane 16-half WMMA fragment from two aligned 16B chunks
__device__ __forceinline__ v16h ld_frag(const _Float16* p) {
  v8h a = *(const v8h*)p;
  v8h b = *(const v8h*)(p + 8);
  v16h r;
#pragma unroll
  for (int i = 0; i < 8; ++i) { r[i] = a[i]; r[i + 8] = b[i]; }
  return r;
}

// ---------- layout converters / weight packers ----------

// NCHW f32 -> NHWC f16
__global__ __launch_bounds__(256) void k_nchw2nhwc(const float* __restrict__ in,
                                                   _Float16* __restrict__ out) {
  int idx = blockIdx.x * 256 + threadIdx.x;            // (b,h,w,c)
  int c = idx & 63, w = (idx >> 6) & 127, h = (idx >> 13) & 127, b = idx >> 20;
  out[idx] = (_Float16)in[(((size_t)b * CH + c) * HH + h) * WW + w];
}

// conv weight (64,64,3,3) f32 -> A-fragment order [mt(4)][tap(9)][kc(2)][lane*16+j] f16
// A packed in contiguous K order used here: K = (lane/16)*16 + j
__global__ __launch_bounds__(256) void k_pack_conv_a(const float* __restrict__ w,
                                                     _Float16* __restrict__ out) {
  int idx = blockIdx.x * 256 + threadIdx.x;
  if (idx >= 4 * 9 * 2 * 512) return;
  int j   = idx & 15;
  int l   = (idx >> 4) & 31;
  int kc  = (idx >> 9) & 1;
  int tap = (idx >> 10) % 9;
  int mt  = idx / (9 * 2 * 512);
  int m = mt * 16 + (l & 15);
  int k = (l >> 4) * 16 + j;
  k += kc * 32;
  int dy = tap / 3, dx = tap % 3;
  out[idx] = (_Float16)w[((m * 64 + k) * 3 + dy) * 3 + dx];
}

// linear weight W[N][K] f32 -> B-fragment order [nt][kc][lane*16+j] f16 (zero padded)
// B 32x16 layout: slot j => k = j<8 ? (lane/16)*8 + j : 16 + (lane/16)*8 + (j-8)
__global__ __launch_bounds__(256) void k_pack_b(const float* __restrict__ w,
                                                _Float16* __restrict__ out,
                                                int Nt, int Kc, int N, int K) {
  int total = Nt * Kc * 512;
  int idx = blockIdx.x * 256 + threadIdx.x;
  if (idx >= total) return;
  int j  = idx & 15;
  int l  = (idx >> 4) & 31;
  int kc = (idx >> 9) % Kc;
  int nt = idx / (512 * Kc);
  int n = nt * 16 + (l & 15);
  int k = kc * 32 + ((j < 8) ? (l >> 4) * 8 + j : 16 + (l >> 4) * 8 + (j - 8));
  out[idx] = (n < N && k < K) ? (_Float16)w[(size_t)n * K + k] : (_Float16)0.f;
}

// ---------- 3x3 conv, implicit GEMM via WMMA f16 ----------
// 64-pixel strip per block; async (ASYNCcnt) halo staging into LDS; each wave
// holds 2 accumulators so every weight A-fragment feeds 2 WMMAs.
__global__ __launch_bounds__(256) void k_conv3x3(const _Float16* __restrict__ in,
                                                 const _Float16* __restrict__ wp,
                                                 const _Float16* __restrict__ resid,
                                                 _Float16* __restrict__ outh,
                                                 float* __restrict__ outf,
                                                 int relu) {
  __shared__ __align__(32) _Float16 lds[3 * 66 * 64];  // rows x cols(66 halo) x ch
  const int b = blockIdx.z, h = blockIdx.y, x0 = blockIdx.x * 64;
  const int tid = threadIdx.x;
  // stage halo tile: 3*66 cols x 64ch in 16B (8-channel) chunks, 1584 tasks
  for (int t = tid; t < 3 * 66 * 8; t += 256) {
    int rc = t >> 3, ck = (t & 7) * 8;
    int r = rc / 66, c = rc % 66;
    int hh = h + r - 1, ww = x0 + c - 1;
    unsigned ldsOff = (unsigned)(size_t)&lds[(r * 66 + c) * 64 + ck];
    if (hh < 0 || hh >= HH || ww < 0 || ww >= WW) {
      _Float16* dst = &lds[(r * 66 + c) * 64 + ck];
#pragma unroll
      for (int i = 0; i < 8; ++i) dst[i] = (_Float16)0.f;
    } else {
      unsigned long long src =
          (unsigned long long)(const void*)(in + (((size_t)b * HH + hh) * WW + ww) * 64 + ck);
      asm volatile("global_load_async_to_lds_b128 %0, %1, off"
                   :: "v"(ldsOff), "v"(src) : "memory");
    }
  }
  asm volatile("s_wait_asynccnt 0x0" ::: "memory");
  __syncthreads();

  const int wave = tid >> 5, lane = tid & 31;
  const int mt = wave >> 1;            // out-channel tile (4)
  const int nt = wave & 1;             // 32-pixel half (2)
  const int nl = lane & 15;            // N within tile
  const int kh = lane >> 4;            // K-high half
  v8f acc0 = {}, acc1 = {};
#pragma unroll
  for (int tap = 0; tap < 9; ++tap) {
    const int dy = tap / 3, dx = tap % 3;
#pragma unroll
    for (int kc = 0; kc < 2; ++kc) {
      v16h a = ld_frag(wp + ((mt * 9 + tap) * 2 + kc) * 512 + lane * 16);
      const _Float16* bb0 = &lds[((dy * 66) + nt * 32 + nl + dx) * 64 + kc * 32 + kh * 16];
      v16h b0 = ld_frag(bb0);
      v16h b1 = ld_frag(bb0 + 16 * 64);   // pixel tile +16 columns
      acc0 = __builtin_amdgcn_wmma_f32_16x16x32_f16(false, a, false, b0,
                                                    (short)0, acc0, false, false);
      acc1 = __builtin_amdgcn_wmma_f32_16x16x32_f16(false, a, false, b1,
                                                    (short)0, acc1, false, false);
    }
  }
#pragma unroll
  for (int t = 0; t < 2; ++t) {
    const v8f& acc = t ? acc1 : acc0;
    const int wx = x0 + nt * 32 + t * 16 + nl;
#pragma unroll
    for (int r = 0; r < 8; ++r) {
      int ch = mt * 16 + r + kh * 8;     // C/D layout: M = r + 8*(lane>=16)
      float v = acc[r];
      if (relu) v = v > 0.f ? v : 0.f;
      if (resid) v += (float)resid[(((size_t)b * HH + h) * WW + wx) * 64 + ch];
      if (outh) outh[(((size_t)b * HH + h) * WW + wx) * 64 + ch] = (_Float16)v;
      if (outf) outf[(((size_t)b * CH + ch) * HH + h) * WW + wx] = v;
    }
  }
}

// ---------- WMMA GEMM: out[rows,Nout] = A[rows,K] * Wpacked ----------
// each wave: 16 rows x NT*16 cols; A-fragment loaded once per K step,
// reused for NT WMMAs (NT accumulators in registers).
template <int NT>
__global__ __launch_bounds__(256) void k_gemm_nt(const _Float16* __restrict__ A,
                                                 const _Float16* __restrict__ Bp,
                                                 float* __restrict__ outf,
                                                 int rows, int K, int Nout) {
  const int wave = threadIdx.x >> 5, lane = threadIdx.x & 31;
  const int tt = blockIdx.x * 8 + wave;      // token tile
  const int nt0 = blockIdx.y * NT;           // first feature tile
  if (tt * 16 >= rows) return;               // wave-uniform
  const int Kc = K >> 5;
  const int row = tt * 16 + (lane & 15);
  const int offA = (lane < 16) ? 0 : 8;      // A-fragment K offset per lane half
  v8f acc[NT];
#pragma unroll
  for (int t = 0; t < NT; ++t) acc[t] = (v8f){};
  for (int kc = 0; kc < Kc; ++kc) {
    const _Float16* ap = A + (size_t)row * K + kc * 32 + offA;
    v8h a0 = *(const v8h*)ap;
    v8h a1 = *(const v8h*)(ap + 16);
    v16h a;
#pragma unroll
    for (int i = 0; i < 8; ++i) { a[i] = a0[i]; a[i + 8] = a1[i]; }
#pragma unroll
    for (int t = 0; t < NT; ++t) {
      v16h bfr = ld_frag(Bp + ((size_t)(nt0 + t) * Kc + kc) * 512 + lane * 16);
      acc[t] = __builtin_amdgcn_wmma_f32_16x16x32_f16(false, a, false, bfr,
                                                      (short)0, acc[t], false, false);
    }
  }
  const int kh = lane >> 4, nl = lane & 15;
#pragma unroll
  for (int t = 0; t < NT; ++t)
#pragma unroll
    for (int r = 0; r < 8; ++r)
      outf[(size_t)(tt * 16 + r + kh * 8) * Nout + (nt0 + t) * 16 + nl] = acc[t][r];
}

// ---------- elementwise / reduction kernels ----------

// 2x2 mean pool (bilinear 128->64 with half-pixel centers) -> tokens (b,L,64) f32
__global__ __launch_bounds__(256) void k_pool(const _Float16* __restrict__ xr,
                                              float* __restrict__ tok) {
  int idx = blockIdx.x * 256 + threadIdx.x;     // (b,h,w,c) at 64x64
  int c = idx & 63, w = (idx >> 6) & 63, h = (idx >> 12) & 63, b = idx >> 18;
  size_t base = (((size_t)b * HH + 2 * h) * WW + 2 * w) * 64 + c;
  float v = (float)xr[base] + (float)xr[base + 64] +
            (float)xr[base + (size_t)WW * 64] + (float)xr[base + (size_t)WW * 64 + 64];
  tok[idx] = 0.25f * v;
}

// layernorm over 64 features, one wave per token, f16 output for GEMM A
__global__ __launch_bounds__(256) void k_layernorm(const float* __restrict__ x,
                                                   const float* __restrict__ g,
                                                   const float* __restrict__ bta,
                                                   _Float16* __restrict__ out) {
  int wave = threadIdx.x >> 5, lane = threadIdx.x & 31;
  int row = blockIdx.x * 8 + wave;
  const float* xp = x + (size_t)row * 64;
  float a = xp[lane], b = xp[lane + 32];
  float s = a + b;
#pragma unroll
  for (int m = 1; m < 32; m <<= 1) s += __shfl_xor(s, m, 32);
  float mu = s * (1.f / 64.f);
  float da = a - mu, db = b - mu;
  float v = da * da + db * db;
#pragma unroll
  for (int m = 1; m < 32; m <<= 1) v += __shfl_xor(v, m, 32);
  float inv = rsqrtf(v * (1.f / 64.f) + 1e-5f);
  out[(size_t)row * 64 + lane]      = (_Float16)(da * inv * g[lane] + bta[lane]);
  out[(size_t)row * 64 + lane + 32] = (_Float16)(db * inv * g[lane + 32] + bta[lane + 32]);
}

// causal depthwise conv1d (DC=4) + bias + silu ; reads xc = xz[:, :128]
__global__ __launch_bounds__(256) void k_conv1d_silu(const float* __restrict__ xz,
                                                     const float* __restrict__ cw,
                                                     const float* __restrict__ cb,
                                                     float* __restrict__ uf,
                                                     _Float16* __restrict__ uh) {
  int idx = blockIdx.x * 256 + threadIdx.x;   // (bl, ch)
  int ch = idx & 127, bl = idx >> 7;
  int b = bl >> 12, l = bl & 4095;
  float acc = cb[ch];
#pragma unroll
  for (int t = 0; t < 4; ++t) {
    int ll = l - 3 + t;
    if (ll >= 0) acc += xz[((size_t)(b * LTOK + ll)) * 256 + ch] * cw[ch * 4 + t];
  }
  float sv = acc / (1.f + __expf(-acc));
  uf[idx] = sv;
  uh[idx] = (_Float16)sv;
}

// dt = softplus(xdbl[:, :4] @ dt_w.T + dt_b)
__global__ __launch_bounds__(256) void k_dtproj(const float* __restrict__ xdbl,
                                                const float* __restrict__ dw,
                                                const float* __restrict__ db,
                                                float* __restrict__ dt) {
  int idx = blockIdx.x * 256 + threadIdx.x;   // (bl, n)
  int n = idx & 127, bl = idx >> 7;
  const float* xd = xdbl + (size_t)bl * 48;
  float acc = db[n];
#pragma unroll
  for (int r = 0; r < 4; ++r) acc += xd[r] * dw[n * 4 + r];
  dt[idx] = (acc > 20.f) ? acc : log1pf(__expf(acc));
}

// selective scan, fused with +u*D and *silu(z); lane = (n pair, s); L serial
__global__ __launch_bounds__(256) void k_scan(const float* __restrict__ dt,
                                              const float* __restrict__ xdbl,
                                              const float* __restrict__ u,
                                              const float* __restrict__ xz,
                                              const float* __restrict__ Alog,
                                              const float* __restrict__ Dp,
                                              _Float16* __restrict__ yh) {
  int b = blockIdx.y;
  int wave = threadIdx.x >> 5, lane = threadIdx.x & 31;
  int n = blockIdx.x * 16 + wave * 2 + (lane >> 4);
  int s = lane & 15;
  float A = -__expf(Alog[n * 16 + s]);
  float Dn = Dp[n];
  float h = 0.f;
  size_t base = (size_t)b * LTOK;
  for (int l = 0; l < LTOK; ++l) {
    size_t bl = base + l;
    float dtv = dt[bl * 128 + n];
    float uv  = u[bl * 128 + n];
    float Bv  = xdbl[bl * 48 + 4 + s];
    float Cv  = xdbl[bl * 48 + 20 + s];
    h = __expf(dtv * A) * h + dtv * Bv * uv;
    float p = h * Cv;
    p += __shfl_xor(p, 1, 32); p += __shfl_xor(p, 2, 32);
    p += __shfl_xor(p, 4, 32); p += __shfl_xor(p, 8, 32);
    if (s == 0) {
      float z = xz[bl * 256 + 128 + n];
      float y = (p + uv * Dn) * (z / (1.f + __expf(-z)));
      yh[bl * 128 + n] = (_Float16)y;
    }
  }
}

// softmax over L per (b, ch)
__global__ __launch_bounds__(256) void k_softmaxL(const float* __restrict__ gf,
                                                  float* __restrict__ gate) {
  int b = blockIdx.x >> 6, ch = blockIdx.x & 63;
  __shared__ float red[256];
  const float* p = gf + (size_t)b * LTOK * 64 + ch;
  float mx = -1e30f;
  for (int l = threadIdx.x; l < LTOK; l += 256) mx = fmaxf(mx, p[(size_t)l * 64]);
  red[threadIdx.x] = mx; __syncthreads();
  for (int st = 128; st > 0; st >>= 1) {
    if (threadIdx.x < st) red[threadIdx.x] = fmaxf(red[threadIdx.x], red[threadIdx.x + st]);
    __syncthreads();
  }
  mx = red[0]; __syncthreads();
  float sum = 0.f;
  for (int l = threadIdx.x; l < LTOK; l += 256) sum += __expf(p[(size_t)l * 64] - mx);
  red[threadIdx.x] = sum; __syncthreads();
  for (int st = 128; st > 0; st >>= 1) {
    if (threadIdx.x < st) red[threadIdx.x] += red[threadIdx.x + st];
    __syncthreads();
  }
  float inv = 1.f / red[0];
  float* q = gate + (size_t)b * LTOK * 64 + ch;
  for (int l = threadIdx.x; l < LTOK; l += 256)
    q[(size_t)l * 64] = __expf(p[(size_t)l * 64] - mx) * inv;
}

// tokens + gate * main
__global__ __launch_bounds__(256) void k_combine(const float* __restrict__ tok,
                                                 const float* __restrict__ gate,
                                                 const float* __restrict__ mn,
                                                 float* __restrict__ out) {
  int idx = blockIdx.x * 256 + threadIdx.x;
  out[idx] = tok[idx] + gate[idx] * mn[idx];
}

// bilinear 64->128 upsample (half-pixel centers, edge clamp) -> NHWC f16
__global__ __launch_bounds__(256) void k_upsample(const float* __restrict__ tok2,
                                                  _Float16* __restrict__ out) {
  int idx = blockIdx.x * 256 + threadIdx.x;   // (b,h,w,c) at 128x128
  int c = idx & 63, w = (idx >> 6) & 127, h = (idx >> 13) & 127, b = idx >> 20;
  float fy = h * 0.5f - 0.25f, fx = w * 0.5f - 0.25f;
  int y0 = (int)floorf(fy), x0 = (int)floorf(fx);
  float wy = fy - (float)y0, wx = fx - (float)x0;
  int y1 = y0 + 1, x1 = x0 + 1;
  y0 = max(0, min(63, y0)); y1 = max(0, min(63, y1));
  x0 = max(0, min(63, x0)); x1 = max(0, min(63, x1));
  const float* src = tok2 + (size_t)b * LTOK * 64 + c;
  float v00 = src[(size_t)(y0 * 64 + x0) * 64];
  float v01 = src[(size_t)(y0 * 64 + x1) * 64];
  float v10 = src[(size_t)(y1 * 64 + x0) * 64];
  float v11 = src[(size_t)(y1 * 64 + x1) * 64];
  float v = (1.f - wy) * ((1.f - wx) * v00 + wx * v01) + wy * ((1.f - wx) * v10 + wx * v11);
  out[idx] = (_Float16)v;
}

// ---------- host orchestration ----------
extern "C" void kernel_launch(void* const* d_in, const int* in_sizes, int n_in,
                              void* d_out, int out_size, void* d_ws, size_t ws_size,
                              hipStream_t stream) {
  const float* x      = (const float*)d_in[0];
  const float* cb_w1  = (const float*)d_in[1];
  const float* cb_w2  = (const float*)d_in[2];
  const float* sm_w1  = (const float*)d_in[3];
  const float* sm_w2  = (const float*)d_in[4];
  const float* ln1_w  = (const float*)d_in[5];
  const float* ln1_b  = (const float*)d_in[6];
  const float* ln2_w  = (const float*)d_in[7];
  const float* ln2_b  = (const float*)d_in[8];
  // m1: 9..17, m2: 18..26 (in_w, conv_w, conv_b, xproj_w, dt_w, dt_b, Alog, D, out_w)

  char* cur = (char*)d_ws;
  auto alloc = [&](size_t bytes) -> char* {
    char* p = cur; cur += (bytes + 255) & ~(size_t)255; return p;
  };
  const size_t IMG = (size_t)B4 * HH * WW * CH;        // 4.19M elems
  _Float16* xh   = (_Float16*)alloc(IMG * 2);
  _Float16* c1h  = (_Float16*)alloc(IMG * 2);
  _Float16* xrh  = (_Float16*)alloc(IMG * 2);
  float*    tok  = (float*)alloc((size_t)BLT * 64 * 4);
  _Float16* lnh  = (_Float16*)alloc((size_t)BLT * 64 * 2);
  float*    xz   = (float*)alloc((size_t)BLT * 256 * 4);
  float*    uf   = (float*)alloc((size_t)BLT * 128 * 4);
  _Float16* uh   = (_Float16*)alloc((size_t)BLT * 128 * 2);
  float*    xdbl = (float*)alloc((size_t)BLT * 48 * 4);
  float*    dtb  = (float*)alloc((size_t)BLT * 128 * 4);
  _Float16* yh   = (_Float16*)alloc((size_t)BLT * 128 * 2);
  float*    mnb  = (float*)alloc((size_t)BLT * 64 * 4);
  float*    gtf  = (float*)alloc((size_t)BLT * 64 * 4);
  float*    gate = (float*)alloc((size_t)BLT * 64 * 4);
  float*    tok2 = (float*)alloc((size_t)BLT * 64 * 4);
  _Float16* uph  = (_Float16*)alloc(IMG * 2);
  _Float16* s1h  = (_Float16*)alloc(IMG * 2);
  _Float16* cw1p = (_Float16*)alloc(36864 * 2);
  _Float16* cw2p = (_Float16*)alloc(36864 * 2);
  _Float16* sw1p = (_Float16*)alloc(36864 * 2);
  _Float16* sw2p = (_Float16*)alloc(36864 * 2);
  _Float16* inp  = (_Float16*)alloc(16 * 2 * 512 * 2);  // in_w pack
  _Float16* xpp  = (_Float16*)alloc(3 * 4 * 512 * 2);   // xproj pack
  _Float16* owp  = (_Float16*)alloc(4 * 4 * 512 * 2);   // out_w pack

  // front conv block
  k_nchw2nhwc<<<IMG / 256, 256, 0, stream>>>(x, xh);
  k_pack_conv_a<<<144, 256, 0, stream>>>(cb_w1, cw1p);
  k_pack_conv_a<<<144, 256, 0, stream>>>(cb_w2, cw2p);
  k_pack_conv_a<<<144, 256, 0, stream>>>(sm_w1, sw1p);
  k_pack_conv_a<<<144, 256, 0, stream>>>(sm_w2, sw2p);
  dim3 cgrid(WW / 64, HH, B4);
  k_conv3x3<<<cgrid, 256, 0, stream>>>(xh, cw1p, nullptr, c1h, nullptr, 1);
  k_conv3x3<<<cgrid, 256, 0, stream>>>(c1h, cw2p, xh, xrh, nullptr, 0);
  k_pool<<<(B4 * 64 * 64 * 64) / 256, 256, 0, stream>>>(xrh, tok);

  // two mamba passes
  for (int p = 0; p < 2; ++p) {
    const float* in_w   = (const float*)d_in[9 + p * 9 + 0];
    const float* conv_w = (const float*)d_in[9 + p * 9 + 1];
    const float* conv_b = (const float*)d_in[9 + p * 9 + 2];
    const float* xpro_w = (const float*)d_in[9 + p * 9 + 3];
    const float* dt_w   = (const float*)d_in[9 + p * 9 + 4];
    const float* dt_b   = (const float*)d_in[9 + p * 9 + 5];
    const float* Alog   = (const float*)d_in[9 + p * 9 + 6];
    const float* Dv     = (const float*)d_in[9 + p * 9 + 7];
    const float* out_w  = (const float*)d_in[9 + p * 9 + 8];
    const float* lw = p == 0 ? ln1_w : ln2_w;
    const float* lb = p == 0 ? ln1_b : ln2_b;
    float* dst = p == 0 ? mnb : gtf;

    k_layernorm<<<BLT / 8, 256, 0, stream>>>(tok, lw, lb, lnh);
    k_pack_b<<<64, 256, 0, stream>>>(in_w, inp, 16, 2, 256, 64);
    k_gemm_nt<4><<<dim3(BLT / 128, 4), 256, 0, stream>>>(lnh, inp, xz, BLT, 64, 256);
    k_conv1d_silu<<<(BLT * 128) / 256, 256, 0, stream>>>(xz, conv_w, conv_b, uf, uh);
    k_pack_b<<<24, 256, 0, stream>>>(xpro_w, xpp, 3, 4, 36, 128);
    k_gemm_nt<3><<<dim3(BLT / 128, 1), 256, 0, stream>>>(uh, xpp, xdbl, BLT, 128, 48);
    k_dtproj<<<(BLT * 128) / 256, 256, 0, stream>>>(xdbl, dt_w, dt_b, dtb);
    k_scan<<<dim3(8, B4), 256, 0, stream>>>(dtb, xdbl, uf, xz, Alog, Dv, yh);
    k_pack_b<<<32, 256, 0, stream>>>(out_w, owp, 4, 4, 64, 128);
    k_gemm_nt<4><<<dim3(BLT / 128, 1), 256, 0, stream>>>(yh, owp, dst, BLT, 128, 64);
  }

  // gate, combine, upsample, tail conv block
  k_softmaxL<<<B4 * 64, 256, 0, stream>>>(gtf, gate);
  k_combine<<<(BLT * 64) / 256, 256, 0, stream>>>(tok, gate, mnb, tok2);
  k_upsample<<<IMG / 256, 256, 0, stream>>>(tok2, uph);
  k_conv3x3<<<cgrid, 256, 0, stream>>>(uph, sw1p, nullptr, s1h, nullptr, 1);
  k_conv3x3<<<cgrid, 256, 0, stream>>>(s1h, sw2p, nullptr, nullptr, (float*)d_out, 0);
}
